// Attention_1_15685220565656
// MI455X (gfx1250) — matrix operand order repaired and checked by reading, not compiled, not run on hardware
//
#include <hip/hip_runtime.h>

#define E 1024
#define BATCH 16
#define BM 128
#define BN 64
#define BK 32
#define LDT 40      // padded LDS row stride (bf16 elems) = 32 + 4 DWORDs pad
#define NT 256

typedef __attribute__((ext_vector_type(16))) __bf16 v16bf;
typedef __attribute__((ext_vector_type(8)))  float  v8f;
typedef __attribute__((ext_vector_type(8)))  __bf16 bf16x8;
typedef __attribute__((ext_vector_type(16))) unsigned short v16u;
typedef __attribute__((ext_vector_type(4)))  unsigned int u32x4;
typedef __attribute__((ext_vector_type(8)))  int i32x8;

// ---------------------------------------------------------------- helpers
static __device__ __forceinline__ v16bf vnegbf(v16bf a) {
  v16u u = __builtin_bit_cast(v16u, a);
  u ^= (unsigned short)0x8000u;           // flip sign bit of every bf16
  return __builtin_bit_cast(v16bf, u);
}

static __device__ __forceinline__ v8f wmma_bf16(v16bf a, v16bf b, v8f c) {
  return __builtin_amdgcn_wmma_f32_16x16x32_bf16(false, a, false, b,
                                                 (short)0, c, false, false);
}

// A fragment (16x32 bf16), ISA 7.12.2 layout. LDS tile row-major, stride LDT.
static __device__ __forceinline__ v16bf frag_a(const __bf16* As, int r0, int lane) {
  const int l = lane & 15, hi = (lane >> 4) & 1;
  const __bf16* p = As + (r0 + l) * LDT;
  v16bf f;
#pragma unroll
  for (int j = 0; j < 8; ++j) f[j] = p[hi * 8 + j];
#pragma unroll
  for (int j = 0; j < 8; ++j) f[j + 8] = p[16 + hi * 8 + j];
  return f;
}

// B fragment (32x16 bf16): LDS tile stored [N][K] row-major, stride LDT.
static __device__ __forceinline__ v16bf frag_b(const __bf16* Bs, int n0, int lane) {
  const int l = lane & 15, hi = (lane >> 4) & 1;
  const __bf16* p = Bs + (n0 + l) * LDT;
  v16bf f;
#pragma unroll
  for (int j = 0; j < 8; ++j) f[j] = p[hi * 8 + j];
#pragma unroll
  for (int j = 0; j < 8; ++j) f[j + 8] = p[16 + hi * 8 + j];
  return f;
}

// -------------------------------------------------- Tensor Data Mover issue
// 2D bf16 tile [rows][BK] from row-major global (row stride E elems) into LDS
// at byte offset lds_off, padded to LDT-elem rows via D# pad fields.
// D# per CDNA5 ISA: group0 (4 SGPRs) + group1 (8 SGPRs).
static __device__ __forceinline__ void tdm_load_tile_bf16(unsigned lds_off,
                                                          const __bf16* gptr,
                                                          unsigned rows) {
  const unsigned long long ga = (unsigned long long)(uintptr_t)gptr;
  u32x4 g0;
  g0[0] = 1u;                                    // count=1, user descriptor
  g0[1] = lds_off;                               // lds_addr (bytes)
  g0[2] = (unsigned)(ga & 0xffffffffu);          // global_addr[31:0]
  g0[3] = (unsigned)((ga >> 32) & 0x01ffffffu)   // global_addr[56:32]
        | (2u << 30);                            // type = 2 ("image")
  const unsigned flags = (1u << 16)              // data_size = 2 bytes
                       | (1u << 20)              // pad_enable
                       | (3u << 22)              // pad_interval: 16 DWORDs (64B row)
                       | (3u << 25);             // pad_amount: 4 DWORDs (8 bf16)
  const unsigned td0 = E, td1 = E;               // tensor dims (OOB never hit)
  const unsigned long long s0 = E, s1 = E;       // strides (data_size units)
  i32x8 g1;
  g1[0] = (int)flags;                            // mask=0, no atomic barrier
  g1[1] = (int)((td0 & 0xffffu) << 16);          // [15:0] atomic addr=0
  g1[2] = (int)((td0 >> 16) | ((td1 & 0xffffu) << 16));
  g1[3] = (int)((td1 >> 16) | ((unsigned)BK << 16));     // tile_dim0 = BK
  g1[4] = (int)(rows);                                   // tile_dim1, tile_dim2=0
  g1[5] = (int)(unsigned)(s0 & 0xffffffffu);             // tensor_dim0_stride lo
  g1[6] = (int)((unsigned)((s0 >> 32) & 0xffffu) | ((unsigned)(s1 & 0xffffu) << 16));
  g1[7] = (int)(unsigned)((s1 >> 16) & 0xffffffffu);
  asm volatile("tensor_load_to_lds %0, %1" :: "s"(g0), "s"(g1) : "memory");
}

static __device__ __forceinline__ unsigned lds_off_of(const void* p) {
  return (unsigned)(uintptr_t)p;   // LDS aperture: addr[31:0] = LDS byte offset
}

// ------------------------------------------ manual B-tile loader (transpose)
// global [K][N] row-major -> LDS [N][K] (stride LDT), with f32->bf16 convert.
template <typename TB>
static __device__ __forceinline__ void load_Btile(__bf16* Bs_r, __bf16* Bs_i,
                                                  const TB* gBr, const TB* gBi,
                                                  int k0, int bn0, int tid) {
#pragma unroll
  for (int it = 0; it < (BK * (BN / 4)) / NT; ++it) {
    const int t = it * NT + tid;
    const int kk = t >> 4, ns = (t & 15) * 4;
    const TB* pr = gBr + (size_t)(k0 + kk) * E + bn0 + ns;
    const TB* pi = gBi + (size_t)(k0 + kk) * E + bn0 + ns;
    TB tr[4], ti[4];
#pragma unroll
    for (int j = 0; j < 4; ++j) { tr[j] = pr[j]; ti[j] = pi[j]; }
#pragma unroll
    for (int j = 0; j < 4; ++j) {
      Bs_r[(ns + j) * LDT + kk] = (__bf16)tr[j];
      Bs_i[(ns + j) * LDT + kk] = (__bf16)ti[j];
    }
  }
  if (k0 + BK < E)
    __builtin_prefetch(gBr + (size_t)(k0 + BK + (tid >> 3)) * E + bn0, 0, 0);
}

// ---------------------------------------------------------------- prep
__global__ __launch_bounds__(NT)
void prep_unitary(const float* __restrict__ u,
                  __bf16* __restrict__ ULr, __bf16* __restrict__ ULi,
                  __bf16* __restrict__ URr, __bf16* __restrict__ URi) {
  const int idx = blockIdx.x * NT + threadIdx.x;       // over E*E
  const int i = idx >> 10, j = idx & (E - 1);
  const float ur = u[(size_t)idx * 2 + 0];
  const float ui = u[(size_t)idx * 2 + 1];
  ULr[idx] = (__bf16)ur;
  ULi[idx] = (__bf16)ui;
  URr[(size_t)j * E + i] = (__bf16)ur;                 // U_r^T
  URi[(size_t)j * E + i] = (__bf16)(-ui);              // -U_i^T
}

// ---------------------------------------------------------------- complex GEMM
// C = A * B (complex), A-tiles via TDM, B-tiles manual (transpose/convert),
// double-buffered.  stride 0 => operand shared across batch.
template <typename TB>
__global__ __launch_bounds__(NT)
void cgemm_cplx(const __bf16* __restrict__ Ar, const __bf16* __restrict__ Ai, size_t sA,
                const TB* __restrict__ Br, const TB* __restrict__ Bi, size_t sB,
                __bf16* __restrict__ Cr, __bf16* __restrict__ Ci, size_t sC) {
  __shared__ __bf16 As_r[2][BM * LDT], As_i[2][BM * LDT];
  __shared__ __bf16 Bs_r[2][BN * LDT], Bs_i[2][BN * LDT];

  const int tid = threadIdx.x;
  const int lane = tid & 31, wave = tid >> 5;
  const int wm = (wave & 3) * 32;   // 4 waves along M
  const int wn = (wave >> 2) * 32;  // 2 waves along N
  const int bm0 = blockIdx.y * BM;
  const int bn0 = blockIdx.x * BN;
  const size_t b = blockIdx.z;

  const __bf16* gAr = Ar + b * sA + (size_t)bm0 * E;
  const __bf16* gAi = Ai + b * sA + (size_t)bm0 * E;
  const TB* gBr = Br + b * sB;
  const TB* gBi = Bi + b * sB;

  v8f accR[2][2] = {}, accI[2][2] = {};

  // ---- stage 0
  if (wave == 0) {
    tdm_load_tile_bf16(lds_off_of(&As_r[0][0]), gAr, BM);
    tdm_load_tile_bf16(lds_off_of(&As_i[0][0]), gAi, BM);
  }
  load_Btile<TB>(Bs_r[0], Bs_i[0], gBr, gBi, 0, bn0, tid);
  if (wave == 0) __builtin_amdgcn_s_wait_tensorcnt(0);
  __syncthreads();

  const int NKT = E / BK;
  for (int kt = 0; kt < NKT; ++kt) {
    const int cur = kt & 1, nxt = cur ^ 1;
    if (kt + 1 < NKT) {                 // prefetch next tile into other buffer
      const int k1 = (kt + 1) * BK;
      if (wave == 0) {
        tdm_load_tile_bf16(lds_off_of(&As_r[nxt][0]), gAr + k1, BM);
        tdm_load_tile_bf16(lds_off_of(&As_i[nxt][0]), gAi + k1, BM);
      }
      load_Btile<TB>(Bs_r[nxt], Bs_i[nxt], gBr, gBi, k1, bn0, tid);
    }

    const v16bf br0 = frag_b(Bs_r[cur], wn +  0, lane);
    const v16bf br1 = frag_b(Bs_r[cur], wn + 16, lane);
    const v16bf bi0 = frag_b(Bs_i[cur], wn +  0, lane);
    const v16bf bi1 = frag_b(Bs_i[cur], wn + 16, lane);
#pragma unroll
    for (int mi = 0; mi < 2; ++mi) {
      const v16bf ar = frag_a(As_r[cur], wm + mi * 16, lane);
      const v16bf ai = frag_a(As_i[cur], wm + mi * 16, lane);
      const v16bf an = vnegbf(ai);                 // C_r = ArBr - AiBi
      accR[mi][0] = wmma_bf16(ar, br0, accR[mi][0]);
      accR[mi][0] = wmma_bf16(an, bi0, accR[mi][0]);
      accI[mi][0] = wmma_bf16(ar, bi0, accI[mi][0]);
      accI[mi][0] = wmma_bf16(ai, br0, accI[mi][0]);
      accR[mi][1] = wmma_bf16(ar, br1, accR[mi][1]);
      accR[mi][1] = wmma_bf16(an, bi1, accR[mi][1]);
      accI[mi][1] = wmma_bf16(ar, bi1, accI[mi][1]);
      accI[mi][1] = wmma_bf16(ai, br1, accI[mi][1]);
    }
    if (wave == 0) __builtin_amdgcn_s_wait_tensorcnt(0);
    __syncthreads();
  }

  __bf16* cR = Cr + b * sC;
  __bf16* cI = Ci + b * sC;
  const int l = lane & 15, hi = (lane >> 4) & 1;
#pragma unroll
  for (int mi = 0; mi < 2; ++mi)
#pragma unroll
    for (int ni = 0; ni < 2; ++ni) {
      const int n = bn0 + wn + ni * 16 + l;
      const int m0 = bm0 + wm + mi * 16 + hi * 8;
#pragma unroll
      for (int r = 0; r < 8; ++r) {
        cR[(size_t)(m0 + r) * E + n] = (__bf16)accR[mi][ni][r];
        cI[(size_t)(m0 + r) * E + n] = (__bf16)accI[mi][ni][r];
      }
    }
}

// ---------------------------------------------------------------- scores
// S[b,m,n] = sum_e qr*kr + qi*ki  + maskbias(n).  NT GEMM: the K operand is
// naturally [N][K] in memory, so BOTH operands stream through the TDM.
__global__ __launch_bounds__(NT)
void scores_kernel(const __bf16* __restrict__ qr, const __bf16* __restrict__ qi,
                   const __bf16* __restrict__ kr, const __bf16* __restrict__ ki,
                   const float* __restrict__ Kmask, float* __restrict__ S) {
  __shared__ __bf16 Qs_r[2][BM * LDT], Qs_i[2][BM * LDT];
  __shared__ __bf16 Ks_r[2][BN * LDT], Ks_i[2][BN * LDT];

  const int tid = threadIdx.x;
  const int lane = tid & 31, wave = tid >> 5;
  const int wm = (wave & 3) * 32, wn = (wave >> 2) * 32;
  const int bm0 = blockIdx.y * BM, bn0 = blockIdx.x * BN;
  const size_t b = blockIdx.z, sE = (size_t)E * E;

  const __bf16* gqr = qr + b * sE + (size_t)bm0 * E;
  const __bf16* gqi = qi + b * sE + (size_t)bm0 * E;
  const __bf16* gkr = kr + b * sE + (size_t)bn0 * E;
  const __bf16* gki = ki + b * sE + (size_t)bn0 * E;

  v8f acc[2][2] = {};

  if (wave == 0) {
    tdm_load_tile_bf16(lds_off_of(&Qs_r[0][0]), gqr, BM);
    tdm_load_tile_bf16(lds_off_of(&Qs_i[0][0]), gqi, BM);
    tdm_load_tile_bf16(lds_off_of(&Ks_r[0][0]), gkr, BN);
    tdm_load_tile_bf16(lds_off_of(&Ks_i[0][0]), gki, BN);
    __builtin_amdgcn_s_wait_tensorcnt(0);
  }
  __syncthreads();

  const int NKT = E / BK;
  for (int kt = 0; kt < NKT; ++kt) {
    const int cur = kt & 1, nxt = cur ^ 1;
    if (kt + 1 < NKT) {
      const int k1 = (kt + 1) * BK;
      if (wave == 0) {
        tdm_load_tile_bf16(lds_off_of(&Qs_r[nxt][0]), gqr + k1, BM);
        tdm_load_tile_bf16(lds_off_of(&Qs_i[nxt][0]), gqi + k1, BM);
        tdm_load_tile_bf16(lds_off_of(&Ks_r[nxt][0]), gkr + k1, BN);
        tdm_load_tile_bf16(lds_off_of(&Ks_i[nxt][0]), gki + k1, BN);
      }
    }

    const v16bf kr0 = frag_b(Ks_r[cur], wn +  0, lane);
    const v16bf kr1 = frag_b(Ks_r[cur], wn + 16, lane);
    const v16bf ki0 = frag_b(Ks_i[cur], wn +  0, lane);
    const v16bf ki1 = frag_b(Ks_i[cur], wn + 16, lane);
#pragma unroll
    for (int mi = 0; mi < 2; ++mi) {
      const v16bf qrf = frag_a(Qs_r[cur], wm + mi * 16, lane);
      const v16bf qif = frag_a(Qs_i[cur], wm + mi * 16, lane);
      acc[mi][0] = wmma_bf16(qrf, kr0, acc[mi][0]);
      acc[mi][0] = wmma_bf16(qif, ki0, acc[mi][0]);
      acc[mi][1] = wmma_bf16(qrf, kr1, acc[mi][1]);
      acc[mi][1] = wmma_bf16(qif, ki1, acc[mi][1]);
    }
    if (wave == 0) __builtin_amdgcn_s_wait_tensorcnt(0);
    __syncthreads();
  }

  float* Sb = S + b * sE;
  const float* km = Kmask + b * E;
  const int l = lane & 15, hi = (lane >> 4) & 1;
#pragma unroll
  for (int mi = 0; mi < 2; ++mi)
#pragma unroll
    for (int ni = 0; ni < 2; ++ni) {
      const int n = bn0 + wn + ni * 16 + l;
      const float bias = (1.0f - km[n]) * -100000.0f;
      const int m0 = bm0 + wm + mi * 16 + hi * 8;
#pragma unroll
      for (int r = 0; r < 8; ++r)
        Sb[(size_t)(m0 + r) * E + n] = acc[mi][ni][r] + bias;
    }
}

// ---------------------------------------------------------------- softmax
__global__ __launch_bounds__(NT)
void softmax_rows(const float* __restrict__ S, __bf16* __restrict__ P) {
  __shared__ float red[NT];
  const int tid = threadIdx.x;
  const size_t row = blockIdx.x;
  const float* s = S + row * E;

  float v[4];
  float mx = -1e30f;
#pragma unroll
  for (int i = 0; i < 4; ++i) { v[i] = s[tid + i * NT]; mx = fmaxf(mx, v[i]); }
  red[tid] = mx; __syncthreads();
  for (int off = NT / 2; off > 0; off >>= 1) {
    if (tid < off) red[tid] = fmaxf(red[tid], red[tid + off]);
    __syncthreads();
  }
  const float m = red[0];
  __syncthreads();

  float sum = 0.f;
#pragma unroll
  for (int i = 0; i < 4; ++i) { v[i] = __expf(v[i] - m); sum += v[i]; }
  red[tid] = sum; __syncthreads();
  for (int off = NT / 2; off > 0; off >>= 1) {
    if (tid < off) red[tid] += red[tid + off];
    __syncthreads();
  }
  const float inv = 1.0f / red[0];
#pragma unroll
  for (int i = 0; i < 4; ++i)
    P[row * E + tid + i * NT] = (__bf16)(v[i] * inv);
}

// ---------------------------------------------------------------- probs @ v
__global__ __launch_bounds__(NT)
void av_kernel(const __bf16* __restrict__ Pp,
               const __bf16* __restrict__ vr, const __bf16* __restrict__ vi,
               const float* __restrict__ Qmask,
               float* __restrict__ Or, float* __restrict__ Oi) {
  __shared__ __bf16 Ps[2][BM * LDT];
  __shared__ __bf16 Vs_r[2][BN * LDT], Vs_i[2][BN * LDT];

  const int tid = threadIdx.x;
  const int lane = tid & 31, wave = tid >> 5;
  const int wm = (wave & 3) * 32, wn = (wave >> 2) * 32;
  const int bm0 = blockIdx.y * BM, bn0 = blockIdx.x * BN;
  const size_t b = blockIdx.z, sE = (size_t)E * E;

  const __bf16* gP = Pp + b * sE + (size_t)bm0 * E;
  const __bf16* gvr = vr + b * sE;
  const __bf16* gvi = vi + b * sE;

  v8f accR[2][2] = {}, accI[2][2] = {};

  if (wave == 0) tdm_load_tile_bf16(lds_off_of(&Ps[0][0]), gP, BM);
  load_Btile<__bf16>(Vs_r[0], Vs_i[0], gvr, gvi, 0, bn0, tid);
  if (wave == 0) __builtin_amdgcn_s_wait_tensorcnt(0);
  __syncthreads();

  const int NKT = E / BK;
  for (int kt = 0; kt < NKT; ++kt) {
    const int cur = kt & 1, nxt = cur ^ 1;
    if (kt + 1 < NKT) {
      const int k1 = (kt + 1) * BK;
      if (wave == 0) tdm_load_tile_bf16(lds_off_of(&Ps[nxt][0]), gP + k1, BM);
      load_Btile<__bf16>(Vs_r[nxt], Vs_i[nxt], gvr, gvi, k1, bn0, tid);
    }

    const v16bf vr0 = frag_b(Vs_r[cur], wn +  0, lane);
    const v16bf vr1 = frag_b(Vs_r[cur], wn + 16, lane);
    const v16bf vi0 = frag_b(Vs_i[cur], wn +  0, lane);
    const v16bf vi1 = frag_b(Vs_i[cur], wn + 16, lane);
#pragma unroll
    for (int mi = 0; mi < 2; ++mi) {
      const v16bf pf = frag_a(Ps[cur], wm + mi * 16, lane);
      accR[mi][0] = wmma_bf16(pf, vr0, accR[mi][0]);
      accI[mi][0] = wmma_bf16(pf, vi0, accI[mi][0]);
      accR[mi][1] = wmma_bf16(pf, vr1, accR[mi][1]);
      accI[mi][1] = wmma_bf16(pf, vi1, accI[mi][1]);
    }
    if (wave == 0) __builtin_amdgcn_s_wait_tensorcnt(0);
    __syncthreads();
  }

  float* oR = Or + b * sE;
  float* oI = Oi + b * sE;
  const float* qm = Qmask + b * E;
  const int l = lane & 15, hi = (lane >> 4) & 1;
#pragma unroll
  for (int mi = 0; mi < 2; ++mi)
#pragma unroll
    for (int ni = 0; ni < 2; ++ni) {
      const int n = bn0 + wn + ni * 16 + l;
      const int m0 = bm0 + wm + mi * 16 + hi * 8;
#pragma unroll
      for (int r = 0; r < 8; ++r) {
        const float g = qm[m0 + r];
        oR[(size_t)(m0 + r) * E + n] = accR[mi][ni][r] * g;
        oI[(size_t)(m0 + r) * E + n] = accI[mi][ni][r] * g;
      }
    }
}

// ---------------------------------------------------------------- launch
extern "C" void kernel_launch(void* const* d_in, const int* in_sizes, int n_in,
                              void* d_out, int out_size, void* d_ws, size_t ws_size,
                              hipStream_t stream) {
  const float* KV_r  = (const float*)d_in[0];
  const float* KV_i  = (const float*)d_in[1];
  const float* Q_r   = (const float*)d_in[2];
  const float* Q_i   = (const float*)d_in[3];
  const float* Kmask = (const float*)d_in[4];
  const float* Qmask = (const float*)d_in[5];
  const float* Uni[3] = { (const float*)d_in[6], (const float*)d_in[7], (const float*)d_in[8] };

  const size_t EE  = (size_t)E * E;           // 1M elems
  const size_t BEE = (size_t)BATCH * EE;      // 16M elems

  // workspace layout (bf16 units); total ~197M bf16 units (~394 MB)
  __bf16* w = (__bf16*)d_ws;
  size_t c = 0;
  __bf16 *ULr[3], *ULi[3], *URr[3], *URi[3];
  for (int u = 0; u < 3; ++u) {
    ULr[u] = w + c; c += EE;
    ULi[u] = w + c; c += EE;
    URr[u] = w + c; c += EE;
    URi[u] = w + c; c += EE;
  }
  __bf16* Pr = w + c; c += BEE;
  __bf16* Pi = w + c; c += BEE;
  __bf16* qrb = w + c; c += BEE;
  __bf16* qib = w + c; c += BEE;
  __bf16* krb = w + c; c += BEE;
  __bf16* kib = w + c; c += BEE;
  __bf16* vrb = w + c; c += BEE;
  __bf16* vib = w + c; c += BEE;
  __bf16* probs = w + c; c += BEE;
  float* Sf = (float*)(w + c);                // BEE floats

  float* out_r = (float*)d_out;
  float* out_i = out_r + BEE;

  for (int u = 0; u < 3; ++u)
    prep_unitary<<<(int)(EE / NT), NT, 0, stream>>>(Uni[u], ULr[u], ULi[u], URr[u], URi[u]);

  const dim3 grid(E / BN, E / BM, BATCH);     // (16, 8, 16)

  // q = Uq @ Q @ Uq^H   (step1 reads f32 inputs directly)
  cgemm_cplx<float ><<<grid, NT, 0, stream>>>(ULr[0], ULi[0], 0, Q_r, Q_i, EE, Pr, Pi, EE);
  cgemm_cplx<__bf16><<<grid, NT, 0, stream>>>(Pr, Pi, EE, URr[0], URi[0], 0, qrb, qib, EE);
  // k = Uk @ KV @ Uk^H
  cgemm_cplx<float ><<<grid, NT, 0, stream>>>(ULr[1], ULi[1], 0, KV_r, KV_i, EE, Pr, Pi, EE);
  cgemm_cplx<__bf16><<<grid, NT, 0, stream>>>(Pr, Pi, EE, URr[1], URi[1], 0, krb, kib, EE);
  // v = Uv @ KV @ Uv^H
  cgemm_cplx<float ><<<grid, NT, 0, stream>>>(ULr[2], ULi[2], 0, KV_r, KV_i, EE, Pr, Pi, EE);
  cgemm_cplx<__bf16><<<grid, NT, 0, stream>>>(Pr, Pi, EE, URr[2], URi[2], 0, vrb, vib, EE);

  scores_kernel<<<grid, NT, 0, stream>>>(qrb, qib, krb, kib, Kmask, Sf);
  softmax_rows<<<BATCH * E, NT, 0, stream>>>(Sf, probs);
  av_kernel<<<grid, NT, 0, stream>>>(probs, vrb, vib, Qmask, out_r, out_i);
}
